// PriorAttention_1597727834687
// MI455X (gfx1250) — matrix operand order repaired and checked by reading, not compiled, not run on hardware
//
#include <hip/hip_runtime.h>
#include <cstddef>

typedef __attribute__((ext_vector_type(16))) __bf16 v16bf;
typedef __attribute__((ext_vector_type(8)))  float  v8f;
typedef __attribute__((ext_vector_type(4)))  unsigned int u32x4;
typedef __attribute__((ext_vector_type(8)))  unsigned int u32x8;

struct alignas(16) US8 { unsigned short u[8]; };

union Frag16 {
  v16bf v;
  unsigned short u[16];
  US8 h[2];
};

static __device__ __forceinline__ unsigned short f2bf(float f) {
  union { float f; unsigned int i; } x; x.f = f;
  unsigned int r = x.i + 0x7FFFu + ((x.i >> 16) & 1u);   // round-to-nearest-even
  return (unsigned short)(r >> 16);
}

// Raw LDS byte offset of a __shared__ object (generic -> AS(3) -> int).
static __device__ __forceinline__ unsigned lds_addr_of(const void* p) {
  return (unsigned)(unsigned long long)(const __attribute__((address_space(3))) void*)p;
}

// ---------------------------------------------------------------------------
// Tensor Data Mover: 2-D tile load (16-bit elements) Global -> LDS.
// D# built per CDNA5 ISA (group0) / §8.4 (group1).
// Issue once per wave; completion tracked with TENSORcnt.
// ---------------------------------------------------------------------------
static __device__ __forceinline__ void tdm_load_2d_b16(
    unsigned long long gaddr,      // global byte address of tile start
    unsigned ldsoff,               // LDS byte offset of destination
    unsigned tile_cols,            // elements per row
    unsigned tile_rows,            // rows
    unsigned row_stride)           // elements between rows in memory
{
  u32x4 g0;
  g0[0] = 1u;                                           // count=1, user mode
  g0[1] = ldsoff;                                       // lds_addr
  g0[2] = (unsigned)(gaddr & 0xFFFFFFFFu);              // global_addr[31:0]
  g0[3] = (unsigned)((gaddr >> 32) & 0x1FFFFFFu)        // global_addr[56:32]
        | (2u << 30);                                   // type = 2 ("image")
  u32x8 g1;
  g1[0] = (1u << 16);                                   // data_size=1 -> 2 bytes
  g1[1] = (tile_cols & 0xFFFFu) << 16;                  // tensor_dim0[15:0]
  g1[2] = ((tile_cols >> 16) & 0xFFFFu)                 // tensor_dim0[31:16]
        | ((tile_rows & 0xFFFFu) << 16);                // tensor_dim1[15:0]
  g1[3] = ((tile_rows >> 16) & 0xFFFFu)                 // tensor_dim1[31:16]
        | ((tile_cols & 0xFFFFu) << 16);                // tile_dim0
  g1[4] = (tile_rows & 0xFFFFu);                        // tile_dim1 (tile_dim2=0)
  g1[5] = row_stride;                                   // tensor_dim0_stride[31:0]
  g1[6] = 0u;                                           // stride[47:32], dim1_stride lo
  g1[7] = 0u;
  asm volatile("tensor_load_to_lds %0, %1" :: "s"(g0), "s"(g1) : "memory");
}

static constexpr int Bb = 8, Ss = 2048, Dd = 512, Hh = 8, Dh = 64;

// ---------------------------------------------------------------------------
// Kernel 1: f32 -> bf16 conversions. x copied straight; weights transposed to
// [N][K] so WMMA B-fragments read 16 consecutive K values per lane.
// ---------------------------------------------------------------------------
__global__ void cvt_kernel(const float* __restrict__ x,
                           const float* __restrict__ Wq, const float* __restrict__ Wk,
                           const float* __restrict__ Wv, const float* __restrict__ Wo,
                           unsigned short* __restrict__ xb,
                           unsigned short* __restrict__ Wtq, unsigned short* __restrict__ Wtk,
                           unsigned short* __restrict__ Wtv, unsigned short* __restrict__ Wto) {
  const long long stride = (long long)gridDim.x * blockDim.x;
  long long tid = (long long)blockIdx.x * blockDim.x + threadIdx.x;
  const long long NX = (long long)Bb * Ss * Dd;
  for (long long i = tid; i < NX; i += stride) xb[i] = f2bf(x[i]);
  const long long NW = (long long)Dd * Dd;
  for (long long i = tid; i < NW; i += stride) {
    int k = (int)(i >> 9), n = (int)(i & 511);
    int t = n * Dd + k;
    Wtq[t] = f2bf(Wq[i]); Wtk[t] = f2bf(Wk[i]);
    Wtv[t] = f2bf(Wv[i]); Wto[t] = f2bf(Wo[i]);
  }
}

// ---------------------------------------------------------------------------
// Kernel 2: projection GEMM  out[b,h,s,dh] = bf16( A[16384x512] @ W + bias ).
// One wave per 16x64 output tile; K-loop step 32; 64 WMMAs per wave.
// ---------------------------------------------------------------------------
__global__ void __launch_bounds__(256) proj_kernel(
    const unsigned short* __restrict__ A,   // [16384][512] bf16, row-major
    const unsigned short* __restrict__ Wt,  // [512 N][512 K] bf16 (transposed)
    const float* __restrict__ bias,         // [512] f32
    unsigned short* __restrict__ outQ)      // [B,H,S,Dh] bf16
{
  const int lane = threadIdx.x & 31;
  const int g = lane >> 4, n16 = lane & 15;
  const int w = blockIdx.x * 8 + (threadIdx.x >> 5);
  const int mt = w >> 3, nt = w & 7;
  const int arow = mt * 16 + n16;

  v8f acc[4] = {};

  for (int k0 = 0; k0 < Dd; k0 += 32) {
    Frag16 a;
    const unsigned short* ap = A + (size_t)arow * Dd + k0;
    a.h[0] = *(const US8*)(ap + g * 16);           // K = k0 + 16g + {0..7}
    a.h[1] = *(const US8*)(ap + g * 16 + 8);       // K = k0 + 16g + 8 + {0..7}
#pragma unroll
    for (int d = 0; d < 4; ++d) {
      Frag16 b;
      const unsigned short* bp = Wt + (size_t)(nt * 64 + d * 16 + n16) * Dd + k0 + g * 16;
      b.h[0] = *(const US8*)(bp);
      b.h[1] = *(const US8*)(bp + 8);
      acc[d] = __builtin_amdgcn_wmma_f32_16x16x32_bf16(
          false, a.v, false, b.v, (short)0, acc[d], false, false);
    }
  }

#pragma unroll
  for (int d = 0; d < 4; ++d) {
    const int c = nt * 64 + d * 16 + n16;
    const float bv = bias[c];
    const int h = c >> 6, dh = c & 63;
#pragma unroll
    for (int r = 0; r < 8; ++r) {
      const int rg = mt * 16 + 8 * g + r;
      const int bi = rg >> 11, s = rg & (Ss - 1);
      outQ[(((size_t)bi * Hh + h) * Ss + s) * Dh + dh] = f2bf(acc[d][r] + bv);
    }
  }
}

// ---------------------------------------------------------------------------
// Kernel 3: flash attention with periodic prior bias.
// Block = 8 waves; each wave owns 16 query rows; key tiles of 32.
// K tile staged via Tensor Data Mover (tensor_load_to_lds + s_wait_tensorcnt);
// V tile staged transposed by the block.
// bias(d) = log1p(2 + exp(-2*(d%30)^2))  (the p=1 term is constant 2).
// ---------------------------------------------------------------------------
static constexpr int KT_OFF = 0;                 // 32*64 u16 (K tile, row-major)
static constexpr int VT_OFF = KT_OFF + 32 * 64;  // 64*32 u16 (V tile, [dh][key])
static constexpr int PT_OFF = VT_OFF + 64 * 32;  // 8*16*32 u16 (per-wave P buffer)
static constexpr int BT_OFF = PT_OFF + 8 * 16 * 32;  // 32 floats (64 u16)
static constexpr int SMEM_U16 = BT_OFF + 64;

__global__ void __launch_bounds__(256) flash_kernel(
    const unsigned short* __restrict__ Q,   // [B,H,S,Dh] bf16
    const unsigned short* __restrict__ K,
    const unsigned short* __restrict__ V,
    unsigned short* __restrict__ O)         // [B,S,D] bf16 (head-merged)
{
  __shared__ unsigned short smem[SMEM_U16];
  unsigned short (*Kt)[64]      = (unsigned short(*)[64])(smem + KT_OFF);
  unsigned short (*Vt)[32]      = (unsigned short(*)[32])(smem + VT_OFF);
  unsigned short (*Pt)[16][32]  = (unsigned short(*)[16][32])(smem + PT_OFF);
  float* biasTab                = (float*)(smem + BT_OFF);

  const int tid = threadIdx.x;
  const int lane = tid & 31, wv = tid >> 5;
  const int g = lane >> 4, n16 = lane & 15;
  const int bh = blockIdx.y;                  // b*H + h
  const int q0 = blockIdx.x * 128 + wv * 16;

  if (tid < 32)
    biasTab[tid] = (tid < 30) ? log1pf(2.0f + __expf(-2.0f * (float)(tid * tid))) : 0.0f;

  const unsigned ldsKt = lds_addr_of(&Kt[0][0]);

  // Q fragments: 16 rows x 64, as two 16x32 A-fragments
  const size_t qbase = ((size_t)bh * Ss + (q0 + n16)) * Dh;
  Frag16 aQ[2];
#pragma unroll
  for (int s = 0; s < 2; ++s) {
    aQ[s].h[0] = *(const US8*)(Q + qbase + s * 32 + g * 16);
    aQ[s].h[1] = *(const US8*)(Q + qbase + s * 32 + g * 16 + 8);
  }

  float mrow[8], lrow[8];
#pragma unroll
  for (int r = 0; r < 8; ++r) { mrow[r] = -1e30f; lrow[r] = 0.0f; }
  v8f acc[4] = {};

  const int li = tid * 8;                     // cooperative copy: 8 bf16 per thread
  const int ckey = li >> 6, cdh = li & 63;

  for (int t0 = 0; t0 < Ss; t0 += 32) {
    __syncthreads();                          // previous tile fully consumed
    // K tile: async DMA via Tensor Data Mover (wave 0 issues; EXEC ignored)
    if (wv == 0) {
      const unsigned long long ga =
          (unsigned long long)(const void*)(K + ((size_t)bh * Ss + t0) * Dh);
      tdm_load_2d_b16(ga, ldsKt, /*cols*/64, /*rows*/32, /*stride*/64);
    }
    // V tile: cooperative load + transposed LDS store
    const size_t kvbase = ((size_t)bh * Ss + t0 + ckey) * Dh + cdh;
    US8 vvv = *(const US8*)(V + kvbase);
    if (t0 + 32 < Ss) __builtin_prefetch(V + kvbase + 32 * Dh, 0, 1);
#pragma unroll
    for (int e = 0; e < 8; ++e) Vt[cdh + e][ckey] = vvv.u[e];
    if (wv == 0) __builtin_amdgcn_s_wait_tensorcnt(0);
    __syncthreads();

    // scores: two 16x16 halves, contraction Dh=64 in two WMMA k-steps
    v8f sc[2] = {};
#pragma unroll
    for (int h = 0; h < 2; ++h) {
#pragma unroll
      for (int s = 0; s < 2; ++s) {
        Frag16 b;
        const unsigned short* bp = &Kt[h * 16 + n16][s * 32 + g * 16];
        b.h[0] = *(const US8*)bp;
        b.h[1] = *(const US8*)(bp + 8);
        sc[h] = __builtin_amdgcn_wmma_f32_16x16x32_bf16(
            false, aQ[s].v, false, b.v, (short)0, sc[h], false, false);
      }
    }

    // scale + prior bias, then online softmax update
    float tmax[8];
#pragma unroll
    for (int r = 0; r < 8; ++r) {
      const int i = q0 + 8 * g + r;
#pragma unroll
      for (int h = 0; h < 2; ++h) {
        const int j = t0 + h * 16 + n16;
        int dist = i - j; if (dist < 0) dist = -dist;
        sc[h][r] = sc[h][r] * 0.125f + biasTab[dist % 30];
      }
      tmax[r] = fmaxf(sc[0][r], sc[1][r]);
#pragma unroll
      for (int off = 1; off < 16; off <<= 1)
        tmax[r] = fmaxf(tmax[r], __shfl_xor(tmax[r], off, 16));
    }

    float alpha[8];
#pragma unroll
    for (int r = 0; r < 8; ++r) {
      const float mn = fmaxf(mrow[r], tmax[r]);
      alpha[r] = __expf(mrow[r] - mn);
      mrow[r] = mn;
      sc[0][r] = __expf(sc[0][r] - mn);
      sc[1][r] = __expf(sc[1][r] - mn);
      float ts = sc[0][r] + sc[1][r];
#pragma unroll
      for (int off = 1; off < 16; off <<= 1)
        ts += __shfl_xor(ts, off, 16);
      lrow[r] = lrow[r] * alpha[r] + ts;
    }
#pragma unroll
    for (int d = 0; d < 4; ++d)
#pragma unroll
      for (int r = 0; r < 8; ++r)
        acc[d][r] *= alpha[r];

    // C-layout -> A-layout transpose of P through wave-private LDS
#pragma unroll
    for (int h = 0; h < 2; ++h)
#pragma unroll
      for (int r = 0; r < 8; ++r)
        Pt[wv][8 * g + r][h * 16 + n16] = f2bf(sc[h][r]);
    // same-wave DS ops are in-order: no barrier needed
    Frag16 aP;
    aP.h[0] = *(const US8*)&Pt[wv][n16][g * 16];
    aP.h[1] = *(const US8*)&Pt[wv][n16][g * 16 + 8];

    // O += P @ V : contraction = 32 keys = one WMMA per 16-wide dh slice
#pragma unroll
    for (int d = 0; d < 4; ++d) {
      Frag16 b;
      const unsigned short* bp = &Vt[d * 16 + n16][g * 16];
      b.h[0] = *(const US8*)bp;
      b.h[1] = *(const US8*)(bp + 8);
      acc[d] = __builtin_amdgcn_wmma_f32_16x16x32_bf16(
          false, aP.v, false, b.v, (short)0, acc[d], false, false);
    }
  }

  // normalize and store head-merged [B,S,D]
  const int b = bh >> 3, h = bh & 7;
#pragma unroll
  for (int r = 0; r < 8; ++r) {
    const float inv = 1.0f / lrow[r];
    const int srow = q0 + 8 * g + r;
#pragma unroll
    for (int d = 0; d < 4; ++d)
      O[((size_t)b * Ss + srow) * Dd + h * Dh + d * 16 + n16] = f2bf(acc[d][r] * inv);
  }
}

// ---------------------------------------------------------------------------
// Kernel 4: output projection, bf16 A @ Wo + bo -> f32 d_out [B,S,D].
// ---------------------------------------------------------------------------
__global__ void __launch_bounds__(256) oproj_kernel(
    const unsigned short* __restrict__ A,   // [16384][512] bf16
    const unsigned short* __restrict__ Wt,  // [512 N][512 K] bf16
    const float* __restrict__ bias,
    float* __restrict__ out)                // [16384][512] f32
{
  const int lane = threadIdx.x & 31;
  const int g = lane >> 4, n16 = lane & 15;
  const int w = blockIdx.x * 8 + (threadIdx.x >> 5);
  const int mt = w >> 3, nt = w & 7;
  const int arow = mt * 16 + n16;

  v8f acc[4] = {};
  for (int k0 = 0; k0 < Dd; k0 += 32) {
    Frag16 a;
    const unsigned short* ap = A + (size_t)arow * Dd + k0;
    a.h[0] = *(const US8*)(ap + g * 16);
    a.h[1] = *(const US8*)(ap + g * 16 + 8);
#pragma unroll
    for (int d = 0; d < 4; ++d) {
      Frag16 b;
      const unsigned short* bp = Wt + (size_t)(nt * 64 + d * 16 + n16) * Dd + k0 + g * 16;
      b.h[0] = *(const US8*)(bp);
      b.h[1] = *(const US8*)(bp + 8);
      acc[d] = __builtin_amdgcn_wmma_f32_16x16x32_bf16(
          false, a.v, false, b.v, (short)0, acc[d], false, false);
    }
  }
#pragma unroll
  for (int d = 0; d < 4; ++d) {
    const int c = nt * 64 + d * 16 + n16;
    const float bv = bias[c];
#pragma unroll
    for (int r = 0; r < 8; ++r) {
      const int rg = mt * 16 + 8 * g + r;
      out[(size_t)rg * Dd + c] = acc[d][r] + bv;
    }
  }
}

// ---------------------------------------------------------------------------
extern "C" void kernel_launch(void* const* d_in, const int* in_sizes, int n_in,
                              void* d_out, int out_size, void* d_ws, size_t ws_size,
                              hipStream_t stream) {
  (void)in_sizes; (void)n_in; (void)out_size;
  const float* x  = (const float*)d_in[0];
  const float* Wq = (const float*)d_in[1];
  const float* bq = (const float*)d_in[2];
  const float* Wk = (const float*)d_in[3];
  const float* bk = (const float*)d_in[4];
  const float* Wv = (const float*)d_in[5];
  const float* bv = (const float*)d_in[6];
  const float* Wo = (const float*)d_in[7];
  const float* bo = (const float*)d_in[8];
  float* out = (float*)d_out;

  const size_t XB = (size_t)Bb * Ss * Dd * sizeof(unsigned short);  // 16 MB
  const size_t WB = (size_t)Dd * Dd * sizeof(unsigned short);       // 0.5 MB
  const size_t need = 5 * XB + 4 * WB;                              // ~86 MB
  if (ws_size < need) return;  // deterministic no-op if scratch too small

  char* ws = (char*)d_ws;
  unsigned short* xb  = (unsigned short*)(ws);
  unsigned short* Wtq = (unsigned short*)(ws + XB);
  unsigned short* Wtk = (unsigned short*)(ws + XB + 1 * WB);
  unsigned short* Wtv = (unsigned short*)(ws + XB + 2 * WB);
  unsigned short* Wto = (unsigned short*)(ws + XB + 3 * WB);
  unsigned short* Qb  = (unsigned short*)(ws + XB + 4 * WB);
  unsigned short* Kb  = (unsigned short*)(ws + 2 * XB + 4 * WB);
  unsigned short* Vb  = (unsigned short*)(ws + 3 * XB + 4 * WB);
  unsigned short* AO  = (unsigned short*)(ws + 4 * XB + 4 * WB);

  cvt_kernel<<<4096, 256, 0, stream>>>(x, Wq, Wk, Wv, Wo, xb, Wtq, Wtk, Wtv, Wto);

  proj_kernel<<<1024, 256, 0, stream>>>(xb, Wtq, bq, Qb);
  proj_kernel<<<1024, 256, 0, stream>>>(xb, Wtk, bk, Kb);
  proj_kernel<<<1024, 256, 0, stream>>>(xb, Wtv, bv, Vb);

  flash_kernel<<<dim3(Ss / 128, Bb * Hh), 256, 0, stream>>>(Qb, Kb, Vb, AO);

  oproj_kernel<<<1024, 256, 0, stream>>>(AO, Wto, bo, out);
}
